// GRCNModel_54434415510100
// MI455X (gfx1250) — compile-verified, matrix operand order checked by reading, not run on hardware
//
#include <hip/hip_runtime.h>
#include <hip/hip_bf16.h>
#include <stdint.h>

#define DIM 64
#define NEG_SLOPE 0.01f

typedef __attribute__((ext_vector_type(16))) __bf16 v16bf;
typedef __attribute__((ext_vector_type(8)))  float  v8f;

struct __align__(16) U4 { unsigned x, y, z, w; };
union BFr { v16bf v; U4 q[2]; };

__device__ __forceinline__ unsigned short f2bf(float f) {
  union { float f; unsigned u; } x; x.f = f;
  unsigned r = x.u + 0x7FFFu + ((x.u >> 16) & 1u);   // round-to-nearest-even
  return (unsigned short)(r >> 16);
}
// pack two floats into packed bf16x2; prefer hw v_cvt_pk_bf16_f32 on gfx1250
__device__ __forceinline__ unsigned pack2bf(float a, float b) {
#if __has_builtin(__builtin_amdgcn_cvt_pk_bf16_f32)
  auto r = __builtin_amdgcn_cvt_pk_bf16_f32(a, b);
  unsigned u; __builtin_memcpy(&u, &r, sizeof(u));
  return u;
#else
  return (unsigned)f2bf(a) | ((unsigned)f2bf(b) << 16);
#endif
}
__device__ __forceinline__ float waveSum(float v) {
#pragma unroll
  for (int m = 16; m > 0; m >>= 1) v += __shfl_xor(v, m, 32);
  return v;
}
// float atomic max valid for mixed signs (init with -inf bits)
__device__ __forceinline__ void atomicMaxF(float* a, float v) {
  if (v >= 0.f) atomicMax((int*)a, __float_as_int(v));
  else          atomicMin((unsigned*)a, __float_as_uint(v));
}

// ---------------- utility fills ----------------
__global__ void k_fill(float* p, long long n, float v) {
  long long i = (long long)blockIdx.x * blockDim.x + threadIdx.x;
  long long s = (long long)gridDim.x * blockDim.x;
  for (; i < n; i += s) p[i] = v;
}
__global__ void k_add(float* x, const float* y, long long n) {
  long long i = (long long)blockIdx.x * blockDim.x + threadIdx.x;
  long long s = (long long)gridDim.x * blockDim.x;
  for (; i < n; i += s) x[i] += y[i];
}

// ---------------- gum = l2norm(Gum), wave per row ----------------
__global__ void k_gum_norm(const float* __restrict__ Gum, float* __restrict__ gum, int NU) {
  int row = blockIdx.x * 8 + (threadIdx.x >> 5);
  int lane = threadIdx.x & 31;
  if (row >= NU) return;
  const float* g = Gum + (long long)row * DIM;
  float a = g[lane], b = g[lane + 32];
  float s = waveSum(a * a + b * b);
  float inv = 1.f / fmaxf(sqrtf(s), 1e-12f);
  float* o = gum + (long long)row * DIM;
  o[lane] = a * inv; o[lane + 32] = b * inv;
}

// ---------------- per-item feat inverse norms (block per row) ----------------
__global__ void k_feat_rnorm(const float* __restrict__ feat, float* __restrict__ rn, int F) {
  int row = blockIdx.x;
  const float* f = feat + (long long)row * F;
  float s = 0.f;
  for (int k = threadIdx.x; k < F; k += blockDim.x) { float v = f[k]; s += v * v; }
  s = waveSum(s);
  __shared__ float p[8];
  if ((threadIdx.x & 31) == 0) p[threadIdx.x >> 5] = s;
  __syncthreads();
  if (threadIdx.x == 0) {
    float t = 0.f;
#pragma unroll
    for (int i = 0; i < 8; ++i) t += p[i];
    rn[row] = 1.f / fmaxf(sqrtf(t), 1e-12f);
  }
}

// ---------------- proj_w [F,64] fp32 -> wT [64,F] bf16 (pairs along k) ----------------
__global__ void k_wtrans(const float* __restrict__ pw, unsigned short* __restrict__ wT, int F) {
  int j = blockIdx.x * blockDim.x + threadIdx.x;      // over (F/2)*64 pairs
  if (j >= (F >> 1) * DIM) return;
  int n = j & 63, k = (j >> 6) << 1;
  unsigned v = pack2bf(pw[(long long)k * DIM + n], pw[(long long)(k + 1) * DIM + n]);
  *(unsigned*)&wT[(long long)n * F + k] = v;
}

// ---------------- WMMA GEMM: gim = l2norm(leaky(l2norm(feat) @ W + b)) ----------------
// block = 128 threads (4 waves). Block computes a 16-row x 64-col strip; each wave
// owns one 16x16 N-tile. K tiled by 32 (bf16 WMMA depth); A tile double-buffered
// in LDS (one barrier per k-step, staging overlaps the WMMAs).
__global__ void __launch_bounds__(128) k_gemm(const float* __restrict__ feat,
                                              const float* __restrict__ rn,
                                              const unsigned short* __restrict__ wT,
                                              const float* __restrict__ bias,
                                              float* __restrict__ gim, int NI, int F) {
  __shared__ __align__(16) unsigned short As[2][16 * 32];  // double-buffered A tile
  __shared__ float Cs[16 * 72];                            // 16x64 (+pad) epilogue tile
  __shared__ float invn[16];
  const int tid = threadIdx.x;
  const int wave = tid >> 5;
  const int lane = tid & 31;
  const int laneL = lane & 15;
  const int hi = lane >> 4;          // half-wave select for fragment K-interleave
  const int row0 = blockIdx.x * 16;
  const int n0 = wave * 16;

  // cooperative A-load mapping: thread loads float4 of feat, scales, converts to bf16
  const int lm = tid >> 3;           // 0..15 row within tile
  const int lk = (tid & 7) * 4;      // k offset within 32
  int srcRow = row0 + lm; if (srcRow >= NI) srcRow = NI - 1;
  const float rscale = rn[srcRow];
  const float* fsrc = feat + (long long)srcRow * F + lk;
  const unsigned short* bbase = wT + (long long)(n0 + laneL) * F + hi * 8;

  auto stage = [&](int k0, int buf) {
    float4 fv = *(const float4*)(fsrc + k0);
    *(unsigned*)&As[buf][lm * 32 + lk]     = pack2bf(fv.x * rscale, fv.y * rscale);
    *(unsigned*)&As[buf][lm * 32 + lk + 2] = pack2bf(fv.z * rscale, fv.w * rscale);
  };

  v8f c = {};
  stage(0, 0);
  __syncthreads();
  for (int k0 = 0; k0 < F; k0 += 32) {
    const int cur = (k0 >> 5) & 1;
    if (k0 + 32 < F) {
      __builtin_prefetch((const void*)(fsrc + k0 + 64), 0, 0);  // global_prefetch_b8
      stage(k0 + 32, cur ^ 1);
    }
    BFr a, b;
    const int ko = hi * 8;
    a.q[0] = *(const U4*)&As[cur][laneL * 32 + ko];        // K {0..7}/{8..15}
    a.q[1] = *(const U4*)&As[cur][laneL * 32 + 16 + ko];   // K {16..23}/{24..31}
    b.q[0] = *(const U4*)(bbase + k0);
    b.q[1] = *(const U4*)(bbase + k0 + 16);
    c = __builtin_amdgcn_wmma_f32_16x16x32_bf16(false, a.v, false, b.v,
                                                (short)0, c, false, false);
    __syncthreads();
  }

  // epilogue: bias + leaky, stage full 16x64 strip, row-l2norm, store
#pragma unroll
  for (int r = 0; r < 8; ++r) {
    int m = r + hi * 8;
    int n = n0 + laneL;
    float v = c[r] + bias[n];
    v = v > 0.f ? v : NEG_SLOPE * v;
    Cs[m * 72 + n] = v;
  }
  __syncthreads();
  if (tid < 16) {
    float s = 0.f;
#pragma unroll
    for (int n = 0; n < DIM; ++n) { float v = Cs[tid * 72 + n]; s += v * v; }
    invn[tid] = 1.f / fmaxf(sqrtf(s), 1e-12f);
  }
  __syncthreads();
  for (int e = tid; e < 16 * DIM; e += 128) {
    int m = e >> 6, n = e & 63;
    int row = row0 + m;
    if (row < NI) gim[(long long)row * DIM + n] = Cs[m * 72 + n] * invn[m];
  }
}

// ---------------- edge attention weight + segment max (wave per edge) ----------------
template <bool DOI>
__global__ void k_edge_w(const float* __restrict__ gum, const float* __restrict__ gim,
                         const int* __restrict__ rows, const int* __restrict__ cols,
                         float* __restrict__ w, float* __restrict__ maxU,
                         float* __restrict__ maxI, int E) {
  int e = blockIdx.x * 8 + (threadIdx.x >> 5);
  int lane = threadIdx.x & 31;
  if (e >= E) return;
  int r = rows[e], cc = cols[e];
  const float* u = gum + (long long)r * DIM;
  const float* v = gim + (long long)cc * DIM;
  float s = waveSum(u[lane] * v[lane] + u[lane + 32] * v[lane + 32]);
  if (lane == 0) {
    float wv = s > 0.f ? s : NEG_SLOPE * s;
    w[e] = wv;
    atomicMaxF(&maxU[r], wv);
    if (DOI) atomicMaxF(&maxI[cc], wv);
  }
}

// ---------------- exp(w - max[seg]) + segment sum (thread per edge) ----------------
__global__ void k_edge_exp(const float* __restrict__ w, const float* __restrict__ mx,
                           const int* __restrict__ seg, float* __restrict__ ebuf,
                           float* __restrict__ sum, int E) {
  int e = blockIdx.x * blockDim.x + threadIdx.x;
  if (e >= E) return;
  int s = seg[e];
  float x = expf(w[e] - mx[s]);
  ebuf[e] = x;
  atomicAdd(&sum[s], x);
}

// ---------------- scatter alpha * src[oth] into dst[seg] (wave per edge) ----------------
__global__ void k_edge_scatter(const float* __restrict__ ebuf, const float* __restrict__ sum,
                               const int* __restrict__ seg, const int* __restrict__ oth,
                               const float* __restrict__ src, float* __restrict__ dst, int E) {
  int e = blockIdx.x * 8 + (threadIdx.x >> 5);
  int lane = threadIdx.x & 31;
  if (e >= E) return;
  int s = seg[e], o = oth[e];
  float alpha = ebuf[e] / (sum[s] + 1e-16f);
  const float* sv = src + (long long)o * DIM;
  float* dv = dst + (long long)s * DIM;
  atomicAdd(&dv[lane], alpha * sv[lane]);
  atomicAdd(&dv[lane + 32], alpha * sv[lane + 32]);
}

// ---------------- gum = l2norm(gum + upd); upd = 0 (wave per row) ----------------
__global__ void k_update_norm(float* __restrict__ gum, float* __restrict__ upd, int NU) {
  int row = blockIdx.x * 8 + (threadIdx.x >> 5);
  int lane = threadIdx.x & 31;
  if (row >= NU) return;
  float* g = gum + (long long)row * DIM;
  float* u = upd + (long long)row * DIM;
  float a = g[lane] + u[lane], b = g[lane + 32] + u[lane + 32];
  u[lane] = 0.f; u[lane + 32] = 0.f;
  float s = waveSum(a * a + b * b);
  float inv = 1.f / fmaxf(sqrtf(s), 1e-12f);
  g[lane] = a * inv; g[lane + 32] = b * inv;
}

// ---------------- out[b] = dot(gu[user[b]], gi[pos[b]]) (wave per pair) ----------------
__global__ void k_final(const float* __restrict__ gu, const float* __restrict__ gi,
                        const int* __restrict__ user, const int* __restrict__ pos,
                        float* __restrict__ out, int B) {
  int b = blockIdx.x * 8 + (threadIdx.x >> 5);
  int lane = threadIdx.x & 31;
  if (b >= B) return;
  const float* u = gu + (long long)user[b] * DIM;
  const float* v = gi + (long long)pos[b] * DIM;
  float s = waveSum(u[lane] * v[lane] + u[lane + 32] * v[lane + 32]);
  if (lane == 0) out[b] = s;
}

extern "C" void kernel_launch(void* const* d_in, const int* in_sizes, int n_in,
                              void* d_out, int out_size, void* d_ws, size_t ws_size,
                              hipStream_t stream) {
  const float* Gum  = (const float*)d_in[0];
  const float* feat = (const float*)d_in[1];
  const float* pw   = (const float*)d_in[2];
  const float* pb   = (const float*)d_in[3];
  const int* rows   = (const int*)d_in[4];
  const int* cols   = (const int*)d_in[5];
  const int* user   = (const int*)d_in[6];
  const int* pos    = (const int*)d_in[7];
  float* out = (float*)d_out;

  const int F  = in_sizes[2] / DIM;
  const int NU = in_sizes[0] / DIM;
  const int NI = in_sizes[1] / F;
  const int E  = in_sizes[4];
  const int B  = in_sizes[6];

  char* p = (char*)d_ws;
  auto carve = [&](size_t bytes) { char* r = p; p += (bytes + 255) & ~(size_t)255; return r; };
  float* gum  = (float*)carve((size_t)NU * DIM * 4);
  float* gim  = (float*)carve((size_t)NI * DIM * 4);
  float* upd  = (float*)carve((size_t)NU * DIM * 4);   // user scatter accumulator
  float* hati = (float*)carve((size_t)NI * DIM * 4);   // item scatter accumulator
  float* rn   = (float*)carve((size_t)NI * 4);
  unsigned short* wT = (unsigned short*)carve((size_t)F * DIM * 2);
  float* w  = (float*)carve((size_t)E * 4);
  float* eu = (float*)carve((size_t)E * 4);
  float* ei = (float*)carve((size_t)E * 4);
  float* mU = (float*)carve((size_t)NU * 4);
  float* sU = (float*)carve((size_t)NU * 4);
  float* mI = (float*)carve((size_t)NI * 4);
  float* sI = (float*)carve((size_t)NI * 4);
  (void)ws_size; (void)n_in; (void)out_size;

  const float NEGINF = -__builtin_huge_valf();
  const int EW = (E + 7) / 8;      // wave-per-edge blocks
  const int ET = (E + 255) / 256;  // thread-per-edge blocks

  k_fill<<<1024, 256, 0, stream>>>(upd, (long long)NU * DIM, 0.f);
  k_gum_norm<<<(NU + 7) / 8, 256, 0, stream>>>(Gum, gum, NU);
  k_feat_rnorm<<<NI, 256, 0, stream>>>(feat, rn, F);
  k_wtrans<<<((F >> 1) * DIM + 255) / 256, 256, 0, stream>>>(pw, wT, F);
  k_gemm<<<(NI + 15) / 16, 128, 0, stream>>>(feat, rn, wT, pb, gim, NI, F);

  for (int it = 0; it < 3; ++it) {
    k_fill<<<256, 256, 0, stream>>>(mU, NU, NEGINF);
    k_fill<<<256, 256, 0, stream>>>(sU, NU, 0.f);
    k_edge_w<false><<<EW, 256, 0, stream>>>(gum, gim, rows, cols, w, mU, nullptr, E);
    k_edge_exp<<<ET, 256, 0, stream>>>(w, mU, rows, eu, sU, E);
    k_edge_scatter<<<EW, 256, 0, stream>>>(eu, sU, rows, cols, gim, upd, E);
    k_update_norm<<<(NU + 7) / 8, 256, 0, stream>>>(gum, upd, NU);  // also re-zeros upd
  }

  // final refining layer (softmax over both user and item segments)
  k_fill<<<256, 256, 0, stream>>>(mU, NU, NEGINF);
  k_fill<<<256, 256, 0, stream>>>(sU, NU, 0.f);
  k_fill<<<256, 256, 0, stream>>>(mI, NI, NEGINF);
  k_fill<<<256, 256, 0, stream>>>(sI, NI, 0.f);
  k_fill<<<1024, 256, 0, stream>>>(hati, (long long)NI * DIM, 0.f);
  k_edge_w<true><<<EW, 256, 0, stream>>>(gum, gim, rows, cols, w, mU, mI, E);
  k_edge_exp<<<ET, 256, 0, stream>>>(w, mU, rows, eu, sU, E);
  k_edge_exp<<<ET, 256, 0, stream>>>(w, mI, cols, ei, sI, E);
  k_edge_scatter<<<EW, 256, 0, stream>>>(eu, sU, rows, cols, gim, upd, E);   // hat_u
  k_edge_scatter<<<EW, 256, 0, stream>>>(ei, sI, cols, rows, gum, hati, E);  // hat_i (reads pre-update gum)
  k_add<<<1024, 256, 0, stream>>>(gum, upd, (long long)NU * DIM);   // gu = gum + hat_u
  k_add<<<1024, 256, 0, stream>>>(gim, hati, (long long)NI * DIM);  // gi = gim + hat_i
  k_final<<<(B + 7) / 8, 256, 0, stream>>>(gum, gim, user, pos, out, B);
}
